// GeoFNO_26405458935886
// MI455X (gfx1250) — compile-verified, matrix operand order checked by reading, not run on hardware
//
#include <hip/hip_runtime.h>
#include <math.h>
#include <stdint.h>

// ---------------- constants ----------------
#define B_   8
#define NPT  16384
#define GH   128
#define GWD  128
#define CW   64            // WIDTH (channels)
#define MD   24            // MODES
#define HM2  48            // 2*MODES kept H rows
#define SPA  (GH*GWD)      // 16384 spatial
#define GRID_ELEMS (B_*CW*SPA)   // 8,388,608
#define NPTS (B_*NPT)            // 131,072
#define NROWS (B_*CW*GH)         // 65,536 (b,c,h) rows
#define TWO_PI_128 0.0490873852123405f

typedef __attribute__((ext_vector_type(16))) _Float16 v16h;
typedef __attribute__((ext_vector_type(8)))  float    v8f;
typedef __attribute__((ext_vector_type(2)))  float    v2f;

__device__ __forceinline__ float gelu_f(float x){
  return 0.5f * x * (1.0f + erff(x * 0.707106781186547524f));
}

// ---------------- WMMA fragment helpers (wave32) --------------
// f16 A (16x32): lane L: M=L&15, kbase=(L>>4)*8; VGPR v<4: K=kbase+2v; v>=4: K=kbase+8+2v
__device__ __forceinline__ v16h frag_a16(const _Float16* s, int ld, int lane){
  int m = lane & 15, kb = (lane >> 4) << 3;
  v16h a;
#pragma unroll
  for (int v = 0; v < 8; ++v){
    int kk = (v < 4) ? (kb + 2*v) : (kb + 8 + 2*v);
    a[2*v]   = s[m*ld + kk];
    a[2*v+1] = s[m*ld + kk + 1];
  }
  return a;
}
// f16 A from k-major f32 tile s[K][16] with on-the-fly conversion
__device__ __forceinline__ v16h frag_a16_kmaj(const float* s, int kbase, int lane){
  int m = lane & 15, kb = (lane >> 4) << 3;
  v16h a;
#pragma unroll
  for (int v = 0; v < 8; ++v){
    int kk = kbase + ((v < 4) ? (kb + 2*v) : (kb + 8 + 2*v));
    a[2*v]   = (_Float16)s[kk*16 + m];
    a[2*v+1] = (_Float16)s[(kk+1)*16 + m];
  }
  return a;
}
// f16 B (32x16): mirrored (N across lanes)
__device__ __forceinline__ v16h frag_b16(const _Float16* s, int ld, int lane){
  int n = lane & 15, kb = (lane >> 4) << 3;
  v16h b;
#pragma unroll
  for (int v = 0; v < 8; ++v){
    int kk = (v < 4) ? (kb + 2*v) : (kb + 8 + 2*v);
    b[2*v]   = s[kk*ld + n];
    b[2*v+1] = s[(kk+1)*ld + n];
  }
  return b;
}
// f32 A (16x4): lane L: M=L&15; VGPR0/1 hold K = (L>=16 ? 2,3 : 0,1)
__device__ __forceinline__ v2f frag_a32(const float* s /*row-major [M][ldK]*/, int ld, int k0, int lane){
  int m = lane & 15;
  int kk = k0 + ((lane >> 4) << 1);
  v2f a; a[0] = s[m*ld + kk]; a[1] = s[m*ld + kk + 1];
  return a;
}
// f32 B (4x16): mirrored
__device__ __forceinline__ v2f frag_b32(const float* s /*[K][ldN]*/, int ld, int k0, int n, int lane){
  int kk = k0 + ((lane >> 4) << 1);
  v2f b; b[0] = s[kk*ld + n]; b[1] = s[(kk+1)*ld + n];
  return b;
}

// ---------------- kernels ----------------
__global__ void k_zero(float* __restrict__ p, int n){
  int i = blockIdx.x * 256 + threadIdx.x;
  if (i < n) p[i] = 0.0f;
}

// precompute twiddle tables: trig[128](cos,sin), TwW[128x48], TiW[48x128]
__global__ __launch_bounds__(256) void k_twiddles(
    float* __restrict__ trig, float* __restrict__ TwW, float* __restrict__ TiW){
  int t = blockIdx.x * 256 + threadIdx.x;   // 0..6143
  if (t < 128){
    trig[2*t]   = cosf(TWO_PI_128 * (float)t);
    trig[2*t+1] = sinf(TWO_PI_128 * (float)t);
  }
  if (t < 128*48){
    // forward W-DFT: Fw[row][2m+p] : p=0 -> +cos, p=1 -> -sin
    int w = t / 48, col = t % 48, m = col >> 1, p = col & 1;
    float th = TWO_PI_128 * (float)((w * m) & 127);
    TwW[t] = p ? -sinf(th) : cosf(th);
    // inverse W-DFT with hermitian fold + 1/128: row k=2m+p, col w
    int k = t / 128, w2 = t % 128; int m2 = k >> 1; int p2 = k & 1;
    float coef = ((m2 == 0) ? 1.0f : 2.0f) * (1.0f/128.0f);
    float th2 = TWO_PI_128 * (float)((m2 * w2) & 127);
    TiW[t] = p2 ? (-coef * sinf(th2)) : (coef * cosf(th2));
  }
}

// deformation MLP + lifting, one thread per point
__global__ __launch_bounds__(256) void k_deform_lift(
    const float* __restrict__ coords, const float* __restrict__ features,
    const float* __restrict__ dw0, const float* __restrict__ db0,
    const float* __restrict__ dw1, const float* __restrict__ db1,
    const float* __restrict__ dw2, const float* __restrict__ db2,
    const float* __restrict__ lw,  const float* __restrict__ lb,
    float* __restrict__ phb, float* __restrict__ pwb, float* __restrict__ feat){
  int p = blockIdx.x * 256 + threadIdx.x;
  if (p >= NPTS) return;
  float c0 = coords[2*p], c1 = coords[2*p+1];
  float h1[32], h2[32];
#pragma unroll
  for (int j = 0; j < 32; ++j)
    h1[j] = gelu_f(db0[j] + c0*dw0[j] + c1*dw0[32+j]);
#pragma unroll
  for (int j = 0; j < 32; ++j){
    float acc = db1[j];
#pragma unroll
    for (int i = 0; i < 32; ++i) acc += h1[i]*dw1[i*32+j];
    h2[j] = gelu_f(acc);
  }
  float d0 = db2[0], d1 = db2[1];
#pragma unroll
  for (int i = 0; i < 32; ++i){ d0 += h2[i]*dw2[2*i]; d1 += h2[i]*dw2[2*i+1]; }
  float l0 = fminf(fmaxf(c0 + d0, -1.0f), 1.0f);
  float l1 = fminf(fmaxf(c1 + d1, -1.0f), 1.0f);
  phb[p] = (l0 + 1.0f) * 0.5f * (float)(GH - 1);
  pwb[p] = (l1 + 1.0f) * 0.5f * (float)(GWD - 1);
  float in5[5] = { features[3*p], features[3*p+1], features[3*p+2], c0, c1 };
#pragma unroll
  for (int o = 0; o < CW; ++o){
    float acc = lb[o];
#pragma unroll
    for (int i = 0; i < 5; ++i) acc += in5[i]*lw[i*CW + o];
    feat[(size_t)p*CW + o] = acc;
  }
}

// P2G scatter-add into (B,HW,C) accumulator + counts
__global__ __launch_bounds__(256) void k_scatter(
    const float* __restrict__ feat, const float* __restrict__ phb,
    const float* __restrict__ pwb, float* __restrict__ acc, float* __restrict__ cnt){
  int p = blockIdx.x * 256 + threadIdx.x;
  if (p >= NPTS) return;
  int b  = p / NPT;
  int ih = min(max((int)rintf(phb[p]), 0), GH - 1);   // round-half-even, like jnp.round
  int iw = min(max((int)rintf(pwb[p]), 0), GWD - 1);
  int cell = b * SPA + ih * GWD + iw;
  atomicAdd(&cnt[cell], 1.0f);
  float* dst = acc + (size_t)cell * CW;
  const float* src = feat + (size_t)p * CW;
  for (int c = 0; c < CW; ++c) atomicAdd(&dst[c], src[c]);
}

// (B,HW,C)/count -> (B,C,H,W)
__global__ __launch_bounds__(256) void k_normtrans(
    const float* __restrict__ acc, const float* __restrict__ cnt, float* __restrict__ g){
  int t = blockIdx.x * 256 + threadIdx.x;
  if (t >= GRID_ELEMS) return;
  int hw = t & (SPA - 1);
  int c  = (t >> 14) & (CW - 1);
  int b  = t >> 20;
  float n = fmaxf(cnt[b * SPA + hw], 1.0f);
  g[t] = acc[(size_t)(b * SPA + hw) * CW + c] / n;
}

// forward W-DFT as f32 WMMA GEMM: Fw(NROWSx48) = X(NROWSx128) * TwW(128x48)
__global__ __launch_bounds__(128) void k_wdft_wmma(
    const float* __restrict__ X, const float* __restrict__ TwW, float* __restrict__ Fw){
  __shared__ __align__(16) float Xs[16 * GWD];   // 16 rows x 128
  __shared__ __align__(16) float Ts[GWD * HM2];  // 128 x 48 twiddles
  const int tid  = threadIdx.x;
  const int row0 = blockIdx.x * 16;
#pragma unroll
  for (int e = 0; e < 16; ++e){ int i = tid + 128*e; Xs[i] = X[(size_t)row0 * GWD + i]; }
#pragma unroll
  for (int e = 0; e < 48; ++e){ int i = tid + 128*e; Ts[i] = TwW[i]; }
  __syncthreads();
  const int wave = tid >> 5, lane = tid & 31;
  if (wave < 3){
    const int n0 = wave * 16;
    const int n  = lane & 15, mb = (lane >> 4) << 3;
    v8f c = {};
#pragma unroll
    for (int ks = 0; ks < 32; ++ks){
      v2f a = frag_a32(Xs, GWD, ks*4, lane);
      v2f b = frag_b32(Ts, HM2, ks*4, n0 + n, lane);
      c = __builtin_amdgcn_wmma_f32_16x16x4_f32(false, a, false, b, (short)0, c, false, false);
    }
#pragma unroll
    for (int r = 0; r < 8; ++r)
      Fw[(size_t)(row0 + mb + r) * HM2 + n0 + n] = c[r];
  }
}

// H-direction complex DFT: 128 rows -> 48 kept rows (0..23 and 104..127)
__global__ __launch_bounds__(256) void k_hdft(
    const float* __restrict__ Fw, const float* __restrict__ trig, float* __restrict__ Fm){
  __shared__ float tl[256];
  tl[threadIdx.x] = trig[threadIdx.x];
  __syncthreads();
  int t = blockIdx.x * 256 + threadIdx.x;           // B*C*48*24 = 589824
  int m  = t % MD;
  int hm = (t / MD) % HM2;
  int bc = t / (MD * HM2);                          // 0..511
  int hp = (hm < MD) ? hm : (80 + hm);
  const float2* in = (const float2*)Fw;
  float ar = 0.0f, ai = 0.0f;
  for (int h = 0; h < GH; ++h){
    int i2 = (h * hp) & 127;
    float c = tl[2*i2], s = -tl[2*i2+1];            // e^{-i theta}
    float2 f = in[((size_t)bc * GH + h) * MD + m];
    ar += f.x * c - f.y * s;
    ai += f.x * s + f.y * c;
  }
  ((float2*)Fm)[((size_t)bc * HM2 + hm) * MD + m] = make_float2(ar, ai);
}

// per-mode complex 64x64 channel mixing
__global__ __launch_bounds__(256) void k_modemix(
    const float* __restrict__ Fm,
    const float* __restrict__ sw1r, const float* __restrict__ sw1i,
    const float* __restrict__ sw2r, const float* __restrict__ sw2i,
    int d, float* __restrict__ Gm){
  int t = blockIdx.x * 256 + threadIdx.x;           // B*64*48*24
  int m  = t % MD;
  int hm = (t / MD) % HM2;
  int o  = (t / (MD * HM2)) % CW;
  int b  = t / (MD * HM2 * CW);
  int x  = (hm < MD) ? hm : (hm - MD);
  const float* wr = (hm < MD) ? sw1r : sw2r;
  const float* wi = (hm < MD) ? sw1i : sw2i;
  size_t wbase = ((size_t)d * CW * CW + o) * (MD * MD) + x * MD + m;
  const float2* in = (const float2*)Fm;
  float ar = 0.0f, ai = 0.0f;
  for (int i = 0; i < CW; ++i){
    float2 xv = in[(((size_t)b * CW + i) * HM2 + hm) * MD + m];
    float wrv = wr[wbase + (size_t)i * CW * MD * MD];
    float wiv = wi[wbase + (size_t)i * CW * MD * MD];
    ar += xv.x * wrv - xv.y * wiv;
    ai += xv.x * wiv + xv.y * wrv;
  }
  ((float2*)Gm)[(((size_t)b * CW + o) * HM2 + hm) * MD + m] = make_float2(ar, ai);
}

// inverse H DFT: 48 kept rows -> 128 rows (with 1/128 normalization)
__global__ __launch_bounds__(256) void k_hidft(
    const float* __restrict__ Gm, const float* __restrict__ trig, float* __restrict__ Gw){
  __shared__ float tl[256];
  tl[threadIdx.x] = trig[threadIdx.x];
  __syncthreads();
  int t = blockIdx.x * 256 + threadIdx.x;           // B*64*128*24
  int m  = t % MD;
  int h  = (t / MD) % GH;
  int bo = t / (MD * GH);                           // 0..511
  const float2* in = (const float2*)Gm;
  float ar = 0.0f, ai = 0.0f;
  for (int hm = 0; hm < HM2; ++hm){
    int hp = (hm < MD) ? hm : (80 + hm);
    int i2 = (hp * h) & 127;
    float c = tl[2*i2], s = tl[2*i2+1];             // e^{+i theta}
    float2 g = in[((size_t)bo * HM2 + hm) * MD + m];
    ar += g.x * c - g.y * s;
    ai += g.x * s + g.y * c;
  }
  ((float2*)Gw)[((size_t)bo * GH + h) * MD + m] =
      make_float2(ar * (1.0f/128.0f), ai * (1.0f/128.0f));
}

// inverse W-DFT + bias as f32 WMMA GEMM: Z(NROWSx128) = Gw(NROWSx48)*TiW(48x128) + bias
__global__ __launch_bounds__(128) void k_widft_wmma(
    const float* __restrict__ Gw, const float* __restrict__ TiW,
    const float* __restrict__ pw_b, int d, float* __restrict__ Z){
  __shared__ __align__(16) float As[16 * HM2];     // 16 rows x 48 (re/im interleaved)
  __shared__ __align__(16) float Ts[HM2 * GWD];    // 48 x 128 twiddles
  const int tid  = threadIdx.x;
  const int row0 = blockIdx.x * 16;                // row = (b*64+o)*128 + h
#pragma unroll
  for (int e = 0; e < 6; ++e){ int i = tid + 128*e; As[i] = Gw[(size_t)row0 * HM2 + i]; }
#pragma unroll
  for (int e = 0; e < 48; ++e){ int i = tid + 128*e; Ts[i] = TiW[i]; }
  __syncthreads();
  const int wave = tid >> 5, lane = tid & 31;
  const int n = lane & 15, mb = (lane >> 4) << 3;
  const int o = (row0 >> 7) & (CW - 1);
  const float bias = pw_b[d * CW + o];
#pragma unroll
  for (int sub = 0; sub < 2; ++sub){
    const int n0 = (wave * 2 + sub) * 16;
    v8f c;
#pragma unroll
    for (int r = 0; r < 8; ++r) c[r] = bias;
#pragma unroll
    for (int ks = 0; ks < 12; ++ks){
      v2f a = frag_a32(As, HM2, ks*4, lane);
      v2f b = frag_b32(Ts, GWD, ks*4, n0 + n, lane);
      c = __builtin_amdgcn_wmma_f32_16x16x4_f32(false, a, false, b, (short)0, c, false, false);
    }
#pragma unroll
    for (int r = 0; r < 8; ++r)
      Z[(size_t)(row0 + mb + r) * GWD + n0 + n] = c[r];
  }
}

// pointwise conv via f16 WMMA: Z[b,o,s] += sum_i X[b,i,s]*W[i,o]
// A tile staged k-major in f32 via async DMA to LDS (ASYNCcnt path)
__global__ __launch_bounds__(128) void k_pwconv(
    const float* __restrict__ X, const float* __restrict__ pw_w,
    float* __restrict__ Z, int d){
  __shared__ __align__(16) float    Asf[CW * 16];  // [k][m] f32
  __shared__ _Float16 Ws[CW * CW];                 // 64 x 64 weights f16
  const int tid = threadIdx.x;
  const int b   = blockIdx.y;
  const int s0  = blockIdx.x * 16;
  const float* Xb = X + (size_t)b * CW * SPA;
  float*       Zb = Z + (size_t)b * CW * SPA;
  const float* Wg = pw_w + d * CW * CW;
#pragma unroll
  for (int e = 0; e < 32; ++e){ int i = tid + 128 * e; Ws[i] = (_Float16)Wg[i]; }
  // async-DMA the 16x64 A tile (both sides 16B-contiguous: 4 spatial floats per op)
#pragma unroll
  for (int e = 0; e < 2; ++e){
    int idx = tid + 128 * e;                        // 0..255
    int k = idx >> 2, m0 = (idx & 3) << 2;
    unsigned int lds = (unsigned int)(uintptr_t)(&Asf[k * 16 + m0]);
    unsigned long long ga =
        (unsigned long long)(uintptr_t)(Xb + (size_t)k * SPA + s0 + m0);
    asm volatile("global_load_async_to_lds_b128 %0, %1, off"
                 :: "v"(lds), "v"(ga) : "memory");
  }
  if (blockIdx.x + 1 < gridDim.x)                   // warm next spatial tile
    __builtin_prefetch(Xb + (size_t)(tid & 63) * SPA + s0 + 16, 0, 1);
  asm volatile("s_wait_asynccnt 0" ::: "memory");
  __syncthreads();
  const int wave = tid >> 5, lane = tid & 31;
  const int o0 = wave * 16;
  const int n  = lane & 15;
  const int mb = (lane >> 4) << 3;
  v8f c;
#pragma unroll
  for (int r = 0; r < 8; ++r)
    c[r] = Zb[(size_t)(o0 + n) * SPA + s0 + (mb + r)];
  v16h a0 = frag_a16_kmaj(Asf, 0,  lane);
  v16h b0 = frag_b16(Ws + o0, CW, lane);
  c = __builtin_amdgcn_wmma_f32_16x16x32_f16(false, a0, false, b0, (short)0, c, false, false);
  v16h a1 = frag_a16_kmaj(Asf, 32, lane);
  v16h b1 = frag_b16(Ws + 32 * CW + o0, CW, lane);
  c = __builtin_amdgcn_wmma_f32_16x16x32_f16(false, a1, false, b1, (short)0, c, false, false);
#pragma unroll
  for (int r = 0; r < 8; ++r)
    Zb[(size_t)(o0 + n) * SPA + s0 + (mb + r)] = c[r];
}

// instance norm stats: one block per (b,o)
__global__ __launch_bounds__(256) void k_stats(const float* __restrict__ Z, float* __restrict__ st){
  __shared__ float ss[256], sq[256];
  int bo = blockIdx.x, tid = threadIdx.x;
  const float* p = Z + (size_t)bo * SPA;
  float s = 0.0f, q = 0.0f;
  for (int e = 0; e < SPA / 256; ++e){
    float v = p[tid + 256 * e];
    s += v; q += v * v;
  }
  ss[tid] = s; sq[tid] = q;
  __syncthreads();
  for (int off = 128; off > 0; off >>= 1){
    if (tid < off){ ss[tid] += ss[tid + off]; sq[tid] += sq[tid + off]; }
    __syncthreads();
  }
  if (tid == 0){
    float mu  = ss[0] / (float)SPA;
    float var = sq[0] / (float)SPA - mu * mu;
    st[2 * bo]     = mu;
    st[2 * bo + 1] = rsqrtf(var + 1e-5f);
  }
}

__global__ __launch_bounds__(256) void k_normgelu(float* __restrict__ Z, const float* __restrict__ st){
  int t = blockIdx.x * 256 + threadIdx.x;
  if (t >= GRID_ELEMS) return;
  int bo = t >> 14;
  float v = (Z[t] - st[2 * bo]) * st[2 * bo + 1];
  Z[t] = gelu_f(v);
}

// fused G2P bilinear gather + p1 GEMM (WMMA) + GELU + p2 dot
__global__ __launch_bounds__(128) void k_proj(
    const float* __restrict__ G, const float* __restrict__ phb, const float* __restrict__ pwb,
    const float* __restrict__ p1w, const float* __restrict__ p1b,
    const float* __restrict__ p2w, const float* __restrict__ p2b,
    float* __restrict__ out){
  __shared__ _Float16 Sh[16 * CW];       // sampled tile (f16)
  __shared__ _Float16 Wh[CW * 128];      // p1 weights
  __shared__ float    Hs[16 * 128];      // hidden tile (f32, post-gelu)
  __shared__ float    tH[16], tW[16];
  __shared__ int      i00[16], i01[16], i10[16], i11[16];
  const int tid = threadIdx.x;
  const int b   = blockIdx.y;
  const int n0  = blockIdx.x * 16;
  const float* Gb = G + (size_t)b * CW * SPA;
#pragma unroll
  for (int e = 0; e < 64; ++e){ int i = tid + 128 * e; Wh[i] = (_Float16)p1w[i]; }
  if (tid < 16){
    int p = b * NPT + n0 + tid;
    float ph = phb[p], pw = pwb[p];
    float h0f = floorf(ph), w0f = floorf(pw);
    tH[tid] = ph - h0f; tW[tid] = pw - w0f;
    int h0 = min(max((int)h0f, 0), GH - 1);
    int w0 = min(max((int)w0f, 0), GWD - 1);
    int h1 = min(h0 + 1, GH - 1);
    int w1 = min(w0 + 1, GWD - 1);
    i00[tid] = h0 * GWD + w0; i01[tid] = h0 * GWD + w1;
    i10[tid] = h1 * GWD + w0; i11[tid] = h1 * GWD + w1;
  }
  __syncthreads();
#pragma unroll
  for (int e = 0; e < 8; ++e){
    int idx = tid + 128 * e;             // 1024 = 16 pts x 64 ch
    int j = idx & 15, cch = idx >> 4;
    const float* gc = Gb + (size_t)cch * SPA;
    float th = tH[j], tw = tW[j];
    float v = (1.0f - th) * (1.0f - tw) * gc[i00[j]]
            + (1.0f - th) * tw          * gc[i01[j]]
            + th          * (1.0f - tw) * gc[i10[j]]
            + th          * tw          * gc[i11[j]];
    Sh[j * CW + cch] = (_Float16)v;
  }
  __syncthreads();
  const int wave = tid >> 5, lane = tid & 31;
  const int n  = lane & 15;
  const int mb = (lane >> 4) << 3;
#pragma unroll
  for (int sub = 0; sub < 2; ++sub){
    int o0 = wave * 32 + sub * 16;
    float bias = p1b[o0 + n];
    v8f c;
#pragma unroll
    for (int r = 0; r < 8; ++r) c[r] = bias;
    v16h a0 = frag_a16(Sh,      CW,  lane);
    v16h b0 = frag_b16(Wh + o0, 128, lane);
    c = __builtin_amdgcn_wmma_f32_16x16x32_f16(false, a0, false, b0, (short)0, c, false, false);
    v16h a1 = frag_a16(Sh + 32,            CW,  lane);
    v16h b1 = frag_b16(Wh + 32 * 128 + o0, 128, lane);
    c = __builtin_amdgcn_wmma_f32_16x16x32_f16(false, a1, false, b1, (short)0, c, false, false);
#pragma unroll
    for (int r = 0; r < 8; ++r)
      Hs[(mb + r) * 128 + o0 + n] = gelu_f(c[r]);
  }
  __syncthreads();
  if (tid < 16){
    float acc = p2b[0];
#pragma unroll 8
    for (int o = 0; o < 128; ++o) acc += Hs[tid * 128 + o] * p2w[o];
    out[(size_t)b * NPT + n0 + tid] = acc;
  }
}

// ---------------- launch ----------------
extern "C" void kernel_launch(void* const* d_in, const int* in_sizes, int n_in,
                              void* d_out, int out_size, void* d_ws, size_t ws_size,
                              hipStream_t stream){
  (void)in_sizes; (void)n_in; (void)out_size; (void)ws_size;
  const float* coords   = (const float*)d_in[0];
  const float* features = (const float*)d_in[1];
  const float* dw0 = (const float*)d_in[2];
  const float* db0 = (const float*)d_in[3];
  const float* dw1 = (const float*)d_in[4];
  const float* db1 = (const float*)d_in[5];
  const float* dw2 = (const float*)d_in[6];
  const float* db2 = (const float*)d_in[7];
  const float* lw  = (const float*)d_in[8];
  const float* lb  = (const float*)d_in[9];
  const float* sw1r = (const float*)d_in[10];
  const float* sw1i = (const float*)d_in[11];
  const float* sw2r = (const float*)d_in[12];
  const float* sw2i = (const float*)d_in[13];
  const float* pw_w = (const float*)d_in[14];
  const float* pw_b = (const float*)d_in[15];
  const float* p1w  = (const float*)d_in[16];
  const float* p1b  = (const float*)d_in[17];
  const float* p2w  = (const float*)d_in[18];
  const float* p2b  = (const float*)d_in[19];
  float* outp = (float*)d_out;

  // workspace layout (floats); Fourier intermediates overlay the feat region
  float* ws    = (float*)d_ws;
  float* gridA = ws;
  float* gridB = gridA + GRID_ELEMS;
  float* feat  = gridB + GRID_ELEMS;
  float* Fw    = feat;                       // NROWS*48 = 3,145,728
  float* Fm    = Fw + 3145728;               // B*C*48*24*2 = 1,179,648
  float* Gm    = Fm + 1179648;               // B*C*48*24*2 = 1,179,648
  float* phb   = feat + GRID_ELEMS;
  float* pwb   = phb + NPTS;
  float* cnt   = pwb + NPTS;
  float* Gw    = cnt + NPTS;                 // NROWS*48 = 3,145,728
  float* stats = Gw + 3145728;               // B*C*2 = 1024
  float* trig  = stats + 1024;               // 128*2
  float* TwW   = trig + 256;                 // 128*48
  float* TiW   = TwW + 6144;                 // 48*128

  // 0) twiddle tables
  k_twiddles<<<24, 256, 0, stream>>>(trig, TwW, TiW);
  // 1) zero accumulators
  k_zero<<<(GRID_ELEMS + 255) / 256, 256, 0, stream>>>(gridB, GRID_ELEMS);
  k_zero<<<(NPTS + 255) / 256, 256, 0, stream>>>(cnt, NPTS);
  // 2) deform + lift
  k_deform_lift<<<NPTS / 256, 256, 0, stream>>>(coords, features, dw0, db0, dw1, db1,
                                                dw2, db2, lw, lb, phb, pwb, feat);
  // 3) P2G scatter-mean, transpose to (B,C,H,W)
  k_scatter<<<NPTS / 256, 256, 0, stream>>>(feat, phb, pwb, gridB, cnt);
  k_normtrans<<<GRID_ELEMS / 256, 256, 0, stream>>>(gridB, cnt, gridA);
  // 4) FNO blocks (ping-pong gridA/gridB)
  for (int d = 0; d < 4; ++d){
    float* X = (d & 1) ? gridB : gridA;
    float* Z = (d & 1) ? gridA : gridB;
    k_wdft_wmma<<<NROWS / 16, 128, 0, stream>>>(X, TwW, Fw);
    k_hdft<<<(B_ * CW * HM2 * MD) / 256, 256, 0, stream>>>(Fw, trig, Fm);
    k_modemix<<<(B_ * CW * HM2 * MD) / 256, 256, 0, stream>>>(Fm, sw1r, sw1i, sw2r, sw2i, d, Gm);
    k_hidft<<<(B_ * CW * GH * MD) / 256, 256, 0, stream>>>(Gm, trig, Gw);
    k_widft_wmma<<<NROWS / 16, 128, 0, stream>>>(Gw, TiW, pw_b, d, Z);
    k_pwconv<<<dim3(SPA / 16, B_), 128, 0, stream>>>(X, pw_w, Z, d);
    k_stats<<<B_ * CW, 256, 0, stream>>>(Z, stats);
    k_normgelu<<<GRID_ELEMS / 256, 256, 0, stream>>>(Z, stats);
  }
  // 5) fused G2P + projection (final grid lives in gridA after d=3)
  k_proj<<<dim3(NPT / 16, B_), 128, 0, stream>>>(gridA, phb, pwb, p1w, p1b, p2w, p2b, outp);
}
